// SimpleRNN_33964601377376
// MI455X (gfx1250) — compile-verified, matrix operand order checked by reading, not run on hardware
//
#include <hip/hip_runtime.h>

typedef __bf16 v16bf __attribute__((ext_vector_type(16)));
typedef __bf16 v8bf  __attribute__((ext_vector_type(8)));
typedef float  v8f   __attribute__((ext_vector_type(8)));
typedef float  v4f   __attribute__((ext_vector_type(4)));

#define TT 512
#define BB 256
#define II 256
#define HH 256
#define GG 768    // 3*H
#define NBLK 32   // persistent blocks; 32 blocks x 8 waves = 256 waves = 256 tiles
#define WROW 264  // padded LDS row (264 bf16 = 528 B = 4 banks offset/row -> conflict-free)

// ---------------- fragment loaders (CDNA5 16-bit A/B 16x32 layout) ----------
// A 16x32 bf16: lane holds row (lane&15); kb = (lane>>4)*8.
// Elements 0..7 -> K=k0+kb+0..7, elements 8..15 -> K=k0+16+kb+0..7.
// B 32x16 assumed symmetric with lane = column n, same K packing, so a B column
// is a contiguous (padded) row of W in LDS.

__device__ __forceinline__ v16bf frag_from_bf16(const __bf16* rowbase, int lane, int k0) {
  const int kb = (lane >> 4) * 8;
  v8bf lo = *(const v8bf*)(rowbase + k0 + kb);
  v8bf hi = *(const v8bf*)(rowbase + k0 + 16 + kb);
  v16bf r;
#pragma unroll
  for (int e = 0; e < 8; ++e) { r[e] = lo[e]; r[e + 8] = hi[e]; }
  return r;
}

__device__ __forceinline__ v16bf frag_from_f32(const float* rowbase, int lane, int k0) {
  const int kb = (lane >> 4) * 8;
  const v4f* p = (const v4f*)(rowbase + k0 + kb);
  const v4f* q = (const v4f*)(rowbase + k0 + 16 + kb);
  v4f a0 = p[0], a1 = p[1], c0 = q[0], c1 = q[1];
  v16bf r;
#pragma unroll
  for (int e = 0; e < 4; ++e) {
    r[e]      = (__bf16)a0[e];
    r[e + 4]  = (__bf16)a1[e];
    r[e + 8]  = (__bf16)c0[e];
    r[e + 12] = (__bf16)c1[e];
  }
  return r;
}

__device__ __forceinline__ float sigmoidf_dev(float v) {
  return 1.0f / (1.0f + __expf(-v));
}
__device__ __forceinline__ float tanhf_dev(float v) {
  return 2.0f * sigmoidf_dev(2.0f * v) - 1.0f;
}

// device-wide step barrier (agent scope)
__device__ __forceinline__ void grid_barrier(unsigned* bar, unsigned target) {
  __threadfence();
  __syncthreads();
  if (threadIdx.x == 0) {
    __hip_atomic_fetch_add(bar, 1u, __ATOMIC_RELEASE, __HIP_MEMORY_SCOPE_AGENT);
    while (__hip_atomic_load(bar, __ATOMIC_ACQUIRE, __HIP_MEMORY_SCOPE_AGENT) < target) {
      __builtin_amdgcn_s_sleep(2);
    }
  }
  __syncthreads();
  __threadfence();
}

// ---------------- init: weights fp32->bf16, zero h0, zero barrier ----------
__global__ void gru_init_kernel(const float* __restrict__ Wih,
                                const float* __restrict__ Whh,
                                __bf16* __restrict__ Wih_bf,
                                __bf16* __restrict__ Whh_bf,
                                float* __restrict__ h_f32,
                                __bf16* __restrict__ h_bf,
                                unsigned* __restrict__ bar) {
  const int idx = blockIdx.x * blockDim.x + threadIdx.x;
  if (idx < GG * II) {
    Wih_bf[idx] = (__bf16)Wih[idx];
    Whh_bf[idx] = (__bf16)Whh[idx];
  }
  if (idx < BB * HH) {
    h_f32[idx] = 0.0f;
    h_bf[idx]  = (__bf16)0.0f;
  }
  if (idx == 0) *bar = 0u;
}

// ---------------- persistent GRU recurrence over all 512 steps -------------
__global__ void __launch_bounds__(256, 1)
gru_persistent_kernel(const float* __restrict__ x,       // [T,B,I]
                      const float* __restrict__ b_ih,    // [3H]
                      const float* __restrict__ b_hh,    // [3H]
                      const __bf16* __restrict__ Wih_bf, // [3H,I]
                      const __bf16* __restrict__ Whh_bf, // [3H,H]
                      float* __restrict__ h_f32,         // [2,B,H] ping-pong
                      __bf16* __restrict__ h_bf,         // [2,B,H] ping-pong
                      unsigned* __restrict__ bar) {
  // 6 gate-matrices x 16 rows x 264 (padded) bf16 = ~49.5 KB LDS, block-shared
  __shared__ __align__(16) __bf16 wlds[6 * 16 * WROW];

  const int lane = threadIdx.x & 31;
  const int wave = threadIdx.x >> 5;
  // waves of a block share nt (-> shared weights in LDS); mt differs per wave
  const int nt = blockIdx.x & 15;                 // hidden-col tile 0..15
  const int mt = ((blockIdx.x >> 4) << 3) + wave; // batch tile 0..15
  const int b0 = mt * 16;
  const int j0 = nt * 16;

  const int n = lane & 15;                  // C/D column within tile
  const int mrow = (lane >> 4) * 8;         // C/D row base within tile
  const int j = j0 + n;                     // global hidden column
  const int arow = b0 + (lane & 15);        // A-fragment row (x and h tiles)

  // ---- stage the block's weight fragments into LDS (t-invariant) ----------
  // gm: 0..2 = W_ih gates r,z,n ; 3..5 = W_hh gates r,z,n
  for (int idx = threadIdx.x; idx < 6 * 16 * 32; idx += 256) {
    const int gm = idx / (16 * 32);
    const int r  = (idx >> 5) & 15;
    const int c  = idx & 31;                           // 8-element chunk
    const int grow = (gm % 3) * HH + j0 + r;           // global weight row
    const __bf16* src =
        (gm < 3 ? Wih_bf + (size_t)grow * II : Whh_bf + (size_t)grow * HH) + c * 8;
    *(v8bf*)(&wlds[gm * 16 * WROW + r * WROW + c * 8]) = *(const v8bf*)src;
  }
  __syncthreads();

  // loop-invariant: biases for this lane's column
  const float bir = b_ih[j],          bhr = b_hh[j];
  const float biz = b_ih[HH + j],     bhz = b_hh[HH + j];
  const float bin = b_ih[2 * HH + j], bhn = b_hh[2 * HH + j];

  // loop-invariant: LDS row bases for the 6 B-fragment sources (lane's column)
  const __bf16* wl_ir = &wlds[0 * 16 * WROW + (lane & 15) * WROW];
  const __bf16* wl_iz = &wlds[1 * 16 * WROW + (lane & 15) * WROW];
  const __bf16* wl_in = &wlds[2 * 16 * WROW + (lane & 15) * WROW];
  const __bf16* wl_hr = &wlds[3 * 16 * WROW + (lane & 15) * WROW];
  const __bf16* wl_hz = &wlds[4 * 16 * WROW + (lane & 15) * WROW];
  const __bf16* wl_hn = &wlds[5 * 16 * WROW + (lane & 15) * WROW];

  for (int t = 0; t < TT; ++t) {
    const float*  x_row   = x + ((size_t)t * BB + arow) * II;
    const __bf16* hin_bf  = h_bf  + (size_t)(t & 1) * BB * HH;
    const float*  hin_f   = h_f32 + (size_t)(t & 1) * BB * HH;
    float*        hout_f  = h_f32 + (size_t)((t + 1) & 1) * BB * HH;
    __bf16*       hout_bf = h_bf  + (size_t)((t + 1) & 1) * BB * HH;
    const __bf16* h_row   = hin_bf + (size_t)arow * HH;

    if (t + 1 < TT) {  // warm L2/L1 for next timestep's x rows
      __builtin_prefetch(x + ((size_t)(t + 1) * BB + arow) * II, 0, 1);
    }

    // per-step: previous h values for this lane's 8 output elements —
    // issue early so the loads hide under the matmul
    float hprev[8];
#pragma unroll
    for (int e = 0; e < 8; ++e) {
      hprev[e] = hin_f[(size_t)(b0 + mrow + e) * HH + j];
    }

    v8f acc_ir = {}, acc_iz = {}, acc_in = {};
    v8f acc_hr = {}, acc_hz = {}, acc_hn = {};

#pragma unroll 4
    for (int k0 = 0; k0 < II; k0 += 32) {
      // gather ALL fragments first (distinct SSA values -> one clause, one wait)
      v16bf ax = frag_from_f32(x_row, lane, k0);   // x tile (fp32 -> bf16)
      v16bf ah = frag_from_bf16(h_row, lane, k0);  // h tile (bf16, global)
      v16bf br = frag_from_bf16(wl_ir, lane, k0);  // 6 B fragments from LDS
      v16bf bz = frag_from_bf16(wl_iz, lane, k0);
      v16bf bn = frag_from_bf16(wl_in, lane, k0);
      v16bf cr = frag_from_bf16(wl_hr, lane, k0);
      v16bf cz = frag_from_bf16(wl_hz, lane, k0);
      v16bf cn = frag_from_bf16(wl_hn, lane, k0);

      // then fire the 6 WMMAs back-to-back
      acc_ir = __builtin_amdgcn_wmma_f32_16x16x32_bf16(false, ax, false, br, (short)0, acc_ir, false, false);
      acc_iz = __builtin_amdgcn_wmma_f32_16x16x32_bf16(false, ax, false, bz, (short)0, acc_iz, false, false);
      acc_in = __builtin_amdgcn_wmma_f32_16x16x32_bf16(false, ax, false, bn, (short)0, acc_in, false, false);
      acc_hr = __builtin_amdgcn_wmma_f32_16x16x32_bf16(false, ah, false, cr, (short)0, acc_hr, false, false);
      acc_hz = __builtin_amdgcn_wmma_f32_16x16x32_bf16(false, ah, false, cz, (short)0, acc_hz, false, false);
      acc_hn = __builtin_amdgcn_wmma_f32_16x16x32_bf16(false, ah, false, cn, (short)0, acc_hn, false, false);
    }

    // gate epilogue: element e of C/D maps to (M = mrow+e, N = n)
#pragma unroll
    for (int e = 0; e < 8; ++e) {
      const int brow = b0 + mrow + e;
      const float r  = sigmoidf_dev(acc_ir[e] + bir + acc_hr[e] + bhr);
      const float z  = sigmoidf_dev(acc_iz[e] + biz + acc_hz[e] + bhz);
      const float nn = tanhf_dev(acc_in[e] + bin + r * (acc_hn[e] + bhn));
      const float hv = (1.0f - z) * nn + z * hprev[e];
      hout_f[(size_t)brow * HH + j]  = hv;
      hout_bf[(size_t)brow * HH + j] = (__bf16)hv;
    }

    grid_barrier(bar, (unsigned)NBLK * (unsigned)(t + 1));
  }
}

// ---------------- output projection: sigmoid(h @ W_outᵀ + b_out) -----------
__global__ void gru_out_kernel(const float* __restrict__ h,      // [B,H] (final)
                               const float* __restrict__ W_out,  // [1,H]
                               const float* __restrict__ b_out,  // [1]
                               float* __restrict__ out) {        // [B,1]
  __shared__ float w[HH];
  w[threadIdx.x] = W_out[threadIdx.x];
  __syncthreads();
  const float* hr = h + (size_t)threadIdx.x * HH;
  float s = 0.0f;
#pragma unroll 8
  for (int k = 0; k < HH; ++k) s += hr[k] * w[k];
  out[threadIdx.x] = sigmoidf_dev(s + b_out[0]);
}

// ---------------- launch ---------------------------------------------------
extern "C" void kernel_launch(void* const* d_in, const int* in_sizes, int n_in,
                              void* d_out, int out_size, void* d_ws, size_t ws_size,
                              hipStream_t stream) {
  const float* x    = (const float*)d_in[0];
  const float* Wih  = (const float*)d_in[1];
  const float* Whh  = (const float*)d_in[2];
  const float* bih  = (const float*)d_in[3];
  const float* bhh  = (const float*)d_in[4];
  const float* Wout = (const float*)d_in[5];
  const float* bout = (const float*)d_in[6];

  char* ws = (char*)d_ws;
  __bf16*   Wih_bf = (__bf16*)(ws);                       // 768*256*2 = 393216 B
  __bf16*   Whh_bf = (__bf16*)(ws + 393216);              // 393216 B
  float*    h_f32  = (float*)(ws + 786432);               // 2*256*256*4 = 524288 B
  __bf16*   h_bf   = (__bf16*)(ws + 1310720);             // 2*256*256*2 = 262144 B
  unsigned* bar    = (unsigned*)(ws + 1572864);           // 4 B

  gru_init_kernel<<<dim3((GG * II + 255) / 256), dim3(256), 0, stream>>>(
      Wih, Whh, Wih_bf, Whh_bf, h_f32, h_bf, bar);

  gru_persistent_kernel<<<dim3(NBLK), dim3(256), 0, stream>>>(
      x, bih, bhh, Wih_bf, Whh_bf, h_f32, h_bf, bar);

  // after 512 steps the final h lives in ping-pong buffer 0
  gru_out_kernel<<<dim3(1), dim3(256), 0, stream>>>(
      h_f32, Wout, bout, (float*)d_out);
}
